// MultiSimilarityLoss2_68702296866888
// MI455X (gfx1250) — matrix operand order repaired and checked by reading, not compiled, not run on hardware
//
#include <hip/hip_runtime.h>
#include <math.h>

typedef __attribute__((ext_vector_type(16))) __bf16 v16bf;
typedef __attribute__((ext_vector_type(8)))  __bf16 v8bf;
typedef __attribute__((ext_vector_type(8)))  float  v8f;
typedef __attribute__((ext_vector_type(4)))  int    v4i;

#define DIM 128          // embedding dim
#define TILE 128         // macro tile (rows == cols per block)
#define KC 32            // K per WMMA / K chunk staged in LDS
#define APITCH 40        // bf16 elems per LDS row: 80B pitch = 16B-aligned, odd slot stride -> optimal banking
#define IGNORE_LABEL 141

#if __has_builtin(__builtin_amdgcn_global_load_async_to_lds_b128)
#define HAVE_ASYNC_LDS 1
typedef __attribute__((address_space(1))) v4i gas_v4i;   // global (device) int4
typedef __attribute__((address_space(3))) v4i las_v4i;   // LDS int4
#else
#define HAVE_ASYNC_LDS 0
#endif

// Assemble a 16-wide bf16 fragment from two contiguous 8-element (16B) LDS chunks.
#define FRAG2(arrp0, arrp1) ({                    \
    union { v16bf v; v8bf h[2]; } _u;             \
    _u.h[0] = *(const v8bf*)(arrp0);              \
    _u.h[1] = *(const v8bf*)(arrp1);              \
    _u.v; })

// ---------------- pre-split: emb (fp32) -> hi/lo bf16 planes (one pass) ----------------
__global__ void ms_split_bf16(const float* __restrict__ emb,
                              unsigned short* __restrict__ hi_u,
                              unsigned short* __restrict__ lo_u, int total)
{
    int idx = blockIdx.x * blockDim.x + threadIdx.x;
    if (idx < total) {
        float x = emb[idx];
        __bf16 h = (__bf16)x;                  // RNE
        __bf16 l = (__bf16)(x - (float)h);     // residual
        ((__bf16*)hi_u)[idx] = h;
        ((__bf16*)lo_u)[idx] = l;
    }
}

// ---------------- fused Gram + multi-similarity reduction ----------------
// Block = 256 threads = 8 wave32. Each wave owns a 16x128 strip of a 128x128
// macro-tile. Split-bf16: acc += Ahi*Bhi + Ahi*Blo + Alo*Bhi  (~fp32 accurate).
__global__ __launch_bounds__(256) void ms_loss_tiles(
    const unsigned short* __restrict__ hi_u, const unsigned short* __restrict__ lo_u,
    const int* __restrict__ gt, float* __restrict__ acc_f, unsigned* __restrict__ acc_u)
{
    const int br = blockIdx.y;
    const int bc = blockIdx.x;
    if (bc < br) return;                  // symmetry: upper-triangle blocks only
    const int rBase = br * TILE;
    const int cBase = bc * TILE;
    const unsigned wgt = (br == bc) ? 1u : 2u;
    const float wf = (float)wgt;

    const __bf16* hi = (const __bf16*)hi_u;
    const __bf16* lo = (const __bf16*)lo_u;

    __shared__ __bf16 Ah[TILE][APITCH];
    __shared__ __bf16 Al[TILE][APITCH];
    __shared__ __bf16 Bh[TILE][APITCH];
    __shared__ __bf16 Bl[TILE][APITCH];

    const int t    = threadIdx.x;
    const int wave = t >> 5;              // 0..7 : row sub-tile of this wave
    const int lane = t & 31;
    const int lrow = lane & 15;
    const int lhi  = lane >> 4;           // 0 or 1

    v8f acc[8];
    #pragma unroll
    for (int nn = 0; nn < 8; ++nn) {
        v8f z = {0.f, 0.f, 0.f, 0.f, 0.f, 0.f, 0.f, 0.f};
        acc[nn] = z;
    }

    const int c  = t & 3;                 // which 16B (8-elem) chunk of the 32-K row
    const int r0 = t >> 2;                // 0..63

    for (int kc = 0; kc < DIM; kc += KC) {
        __syncthreads();
#if HAVE_ASYNC_LDS
        // CDNA5 async copy: global -> LDS directly, no VGPR round trip (ASYNCcnt)
        #pragma unroll
        for (int p = 0; p < 2; ++p) {
            const int row = p * 64 + r0;
            const size_t offA = (size_t)(rBase + row) * DIM + kc + c * 8;
            const size_t offB = (size_t)(cBase + row) * DIM + kc + c * 8;
            __builtin_amdgcn_global_load_async_to_lds_b128(
                (gas_v4i*)(hi + offA), (las_v4i*)&Ah[row][c * 8], 0, 0);
            __builtin_amdgcn_global_load_async_to_lds_b128(
                (gas_v4i*)(lo + offA), (las_v4i*)&Al[row][c * 8], 0, 0);
            __builtin_amdgcn_global_load_async_to_lds_b128(
                (gas_v4i*)(hi + offB), (las_v4i*)&Bh[row][c * 8], 0, 0);
            __builtin_amdgcn_global_load_async_to_lds_b128(
                (gas_v4i*)(lo + offB), (las_v4i*)&Bl[row][c * 8], 0, 0);
        }
#if __has_builtin(__builtin_amdgcn_s_wait_asynccnt)
        __builtin_amdgcn_s_wait_asynccnt(0);
#else
        asm volatile("s_wait_asynccnt 0x0" ::: "memory");
#endif
#else
        // fallback: batch all global loads (8x b128 in flight), then LDS stores
        uint4 vah[2], vAl[2], vbh[2], vbl[2];
        #pragma unroll
        for (int p = 0; p < 2; ++p) {
            const int row = p * 64 + r0;
            const size_t offA = (size_t)(rBase + row) * DIM + kc + c * 8;
            const size_t offB = (size_t)(cBase + row) * DIM + kc + c * 8;
            vah[p] = *(const uint4*)(hi + offA);
            vAl[p] = *(const uint4*)(lo + offA);
            vbh[p] = *(const uint4*)(hi + offB);
            vbl[p] = *(const uint4*)(lo + offB);
        }
        #pragma unroll
        for (int p = 0; p < 2; ++p) {
            const int row = p * 64 + r0;
            *(uint4*)&Ah[row][c * 8] = vah[p];
            *(uint4*)&Al[row][c * 8] = vAl[p];
            *(uint4*)&Bh[row][c * 8] = vbh[p];
            *(uint4*)&Bl[row][c * 8] = vbl[p];
        }
#endif
        __syncthreads();

        // A fragment (16-bit A 16x32): lane holds K[16*lhi..+15], contiguous (matches B)
        const int arow = wave * 16 + lrow;
        v16bf ah = FRAG2(&Ah[arow][16 * lhi], &Ah[arow][16 * lhi + 8]);
        v16bf al = FRAG2(&Al[arow][16 * lhi], &Al[arow][16 * lhi + 8]);

        // B fragment (16-bit B 32x16): lane holds K[16*lhi..+15], contiguous (matches A)
        v16bf bh = FRAG2(&Bh[lrow][16 * lhi], &Bh[lrow][16 * lhi + 8]);
        v16bf bl = FRAG2(&Bl[lrow][16 * lhi], &Bl[lrow][16 * lhi + 8]);

        #pragma unroll
        for (int nn = 0; nn < 8; ++nn) {
            v16bf bh_n = bh, bl_n = bl;
            if (nn < 7) {                 // software-pipeline next column tile's frags
                const int browN = (nn + 1) * 16 + lrow;
                bh_n = FRAG2(&Bh[browN][16 * lhi], &Bh[browN][16 * lhi + 8]);
                bl_n = FRAG2(&Bl[browN][16 * lhi], &Bl[browN][16 * lhi + 8]);
            }
            acc[nn] = __builtin_amdgcn_wmma_f32_16x16x32_bf16(
                false, ah, false, bh, (short)0, acc[nn], false, false);
            acc[nn] = __builtin_amdgcn_wmma_f32_16x16x32_bf16(
                false, ah, false, bl, (short)0, acc[nn], false, false);
            acc[nn] = __builtin_amdgcn_wmma_f32_16x16x32_bf16(
                false, al, false, bh, (short)0, acc[nn], false, false);
            bh = bh_n; bl = bl_n;
        }
    }

    // ---- fused epilogue: masks + partial sums, no matrix materialization ----
    int li[8];
    #pragma unroll
    for (int v = 0; v < 8; ++v)
        li[v] = gt[rBase + wave * 16 + v + 8 * lhi];   // C layout: M = v + 8*(lane/16)

    float sp = 0.f, sn = 0.f;
    unsigned cp = 0u, cn = 0u;
    #pragma unroll
    for (int nn = 0; nn < 8; ++nn) {
        const int j  = cBase + nn * 16 + lrow;          // C layout: N = lane%16
        const int lj = gt[j];
        #pragma unroll
        for (int v = 0; v < 8; ++v) {
            const float s = acc[nn][v];
            const int   i = rBase + wave * 16 + v + 8 * lhi;
            const bool same = (li[v] == lj) && (li[v] != IGNORE_LABEL);
            if (s > 0.f) {
                if (same) {                               // rare (~0.7%): branch pays off
                    if (i != j) { cp += wgt; sp += wf * __expf(-2.0f * (s - 0.5f)); }
                } else {
                    cn += wgt; sn += wf * __expf(40.0f * (s - 0.5f));
                }
            }
        }
    }

    // wave32 butterfly reduction, one atomic per wave per accumulator
    #pragma unroll
    for (int off = 16; off > 0; off >>= 1) {
        sp += __shfl_xor(sp, off, 32);
        sn += __shfl_xor(sn, off, 32);
        cp += __shfl_xor(cp, off, 32);
        cn += __shfl_xor(cn, off, 32);
    }
    if (lane == 0) {
        atomicAdd(&acc_f[0], sp);
        atomicAdd(&acc_f[1], sn);
        atomicAdd(&acc_u[0], cp);
        atomicAdd(&acc_u[1], cn);
    }
}

__global__ void ms_loss_init(float* acc_f, unsigned* acc_u) {
    acc_f[0] = 0.f;
    acc_f[1] = 0.f;
    acc_u[0] = 0u;
    acc_u[1] = 0u;
}

__global__ void ms_loss_final(const float* __restrict__ acc_f,
                              const unsigned* __restrict__ acc_u,
                              float* __restrict__ out) {
    const float sp = acc_f[0], sn = acc_f[1];
    const unsigned cp = acc_u[0], cn = acc_u[1];
    const float pos = (cp > 0u) ? (0.5f   * log1pf(sp) / (float)cp) : 0.f; // 1/SCALE_POS
    const float neg = (cn > 0u) ? (0.025f * log1pf(sn) / (float)cn) : 0.f; // 1/SCALE_NEG
    out[0] = pos + neg;
}

extern "C" void kernel_launch(void* const* d_in, const int* in_sizes, int n_in,
                              void* d_out, int out_size, void* d_ws, size_t ws_size,
                              hipStream_t stream) {
    const float* emb = (const float*)d_in[0];
    const int*   gt  = (const int*)d_in[1];
    const int n = in_sizes[1];                       // 8192 rows

    // ws layout: [0,8) float sums | [8,16) uint counts | [256,...) bf16 hi/lo planes
    float*          acc_f = (float*)d_ws;
    unsigned*       acc_u = (unsigned*)((char*)d_ws + 2 * sizeof(float));
    unsigned short* hi_u  = (unsigned short*)((char*)d_ws + 256);
    unsigned short* lo_u  = hi_u + (size_t)n * DIM;

    const int total = n * DIM;
    ms_split_bf16<<<(total + 255) / 256, 256, 0, stream>>>(emb, hi_u, lo_u, total);

    ms_loss_init<<<1, 1, 0, stream>>>(acc_f, acc_u);

    dim3 grid(n / TILE, n / TILE);
    ms_loss_tiles<<<grid, 256, 0, stream>>>(hi_u, lo_u, gt, acc_f, acc_u);

    ms_loss_final<<<1, 1, 0, stream>>>(acc_f, acc_u, (float*)d_out);
}
